// GroupedQueryAttention_84490596647347
// MI455X (gfx1250) — matrix operand order repaired and checked by reading, not compiled, not run on hardware
//
#include <hip/hip_runtime.h>
#include <hip/hip_bf16.h>
#include <math.h>

// ---------------- problem constants (from reference) ----------------
constexpr int   Bc    = 2;
constexpr int   Nc    = 2048;
constexpr int   DIMc  = 2048;
constexpr int   Hc    = 16;
constexpr int   KVHc  = 4;
constexpr int   HDc   = 128;
constexpr int   NREP  = Hc / KVHc;
constexpr float SCALEc = 0.08838834764831845f; // HD^-0.5

typedef __attribute__((ext_vector_type(16))) __bf16        v16bf;
typedef __attribute__((ext_vector_type(8)))  float          v8f;
typedef __attribute__((ext_vector_type(8)))  unsigned short v8u;
typedef __attribute__((ext_vector_type(4)))  unsigned int   v4u32;
typedef __attribute__((ext_vector_type(8)))  int            v8i32;
typedef __attribute__((ext_vector_type(4)))  int            v4i32;

union FragAB { v16bf bf; v8u h[2]; };

static __device__ __forceinline__ unsigned short f2bf(float f) {
  unsigned int u = __builtin_bit_cast(unsigned int, f);
  u += 0x7FFFu + ((u >> 16) & 1u);           // round-to-nearest-even
  return (unsigned short)(u >> 16);
}

// ---- TDM: issue a 2D bf16 tile load (tile_rows x tile_k elems) into LDS ----
// Descriptor layout per CDNA5 ISA (D# groups 0/1).
// clang-23 toolchain: 6-arg builtin (g0, g1, g2, g3, g4, cpol).
static __device__ __forceinline__ void tdm_load_tile_2d(
    unsigned lds_off_bytes, const void* gaddr,
    unsigned tile_k, unsigned tile_rows,
    unsigned row_stride_elems, unsigned tensor_rows, unsigned tensor_k) {
  unsigned long long ga = (unsigned long long)gaddr;
  v4u32 g0;
  g0[0] = 1u;                                         // count=1, user-mode D#
  g0[1] = lds_off_bytes;                              // lds_addr
  g0[2] = (unsigned)(ga & 0xFFFFFFFFu);               // global_addr[31:0]
  g0[3] = (unsigned)((ga >> 32) & 0x1FFFFFFu)         // global_addr[56:32]
        | (2u << 30);                                 // type=2 ("image")
  const unsigned td0 = tensor_k;                      // tensor_dim0 (elements)
  const unsigned td1 = tensor_rows;                   // tensor_dim1
  const unsigned long long s0 = row_stride_elems;     // tensor_dim0_stride (48b)
  v8i32 g1;
  g1[0] = (int)(1u << 16);                            // wg_mask=0, data_size=1 (2B)
  g1[1] = (int)((td0 & 0xFFFFu) << 16);               // [31:16]=tensor_dim0 lo
  g1[2] = (int)(((td0 >> 16) & 0xFFFFu) | ((td1 & 0xFFFFu) << 16));
  g1[3] = (int)(((td1 >> 16) & 0xFFFFu) | ((tile_k & 0xFFFFu) << 16)); // tile_dim0
  g1[4] = (int)(tile_rows & 0xFFFFu);                 // tile_dim1, tile_dim2=0
  g1[5] = (int)(s0 & 0xFFFFFFFFu);                    // dim0_stride lo
  g1[6] = (int)((s0 >> 32) & 0xFFFFu);                // dim0_stride hi, dim1_stride=0
  g1[7] = 0;
  v4i32 z4 = {0, 0, 0, 0};
  v8i32 z8 = {0, 0, 0, 0, 0, 0, 0, 0};
  __builtin_amdgcn_tensor_load_to_lds(g0, g1, z4, z4, z8, 0);
}

// ---------------- conversion kernels ----------------
__global__ void f32_to_bf16_kernel(const float* __restrict__ in,
                                   unsigned short* __restrict__ out, size_t n) {
  size_t i = (size_t)blockIdx.x * blockDim.x + threadIdx.x;
  if (i < n) out[i] = f2bf(in[i]);
}

// W[k][n] (row-major [Kdim,Nout]) -> WT[n][k] bf16
__global__ void transpose_f32_to_bf16_kernel(const float* __restrict__ W,
                                             unsigned short* __restrict__ WT,
                                             int Kdim, int Nout) {
  size_t i = (size_t)blockIdx.x * blockDim.x + threadIdx.x;
  if (i < (size_t)Kdim * Nout) {
    int k = (int)(i / Nout);
    int n = (int)(i % Nout);
    WT[(size_t)n * Kdim + k] = f2bf(W[i]);
  }
}

// ---------------- LDS-tiled WMMA GEMM: C[M,Nout] = A[M,K] * BT[Nout,K]^T ----
// Block = 256 threads (8 waves) computing a 128x128 output tile.
// A/B 128x32 bf16 tiles are staged in LDS by the Tensor Data Mover,
// double-buffered; each wave owns a 32x64 register tile (2x4 WMMA accums).
// mode 0: bf16 -> Q layout  [B][H][N][HD]
// mode 1: bf16 -> K layout  [B][KVH][N][HD]
// mode 2: bf16 -> VT layout [B][KVH][HD][N]
// mode 3: f32 row-major [M][Nout]
__global__ void __launch_bounds__(256)
wmma_gemm_kernel(const unsigned short* __restrict__ A,
                 const unsigned short* __restrict__ BT,
                 void* __restrict__ out,
                 int NB, int Kdim, int mode) {
  __shared__ unsigned short lsA[2][128 * 32];
  __shared__ unsigned short lsB[2][128 * 32];

  const int lane  = threadIdx.x & 31;
  const int half  = lane >> 4;
  const int l16   = lane & 15;
  const int wave  = threadIdx.x >> 5;
  const int waveM = wave >> 1;          // 0..3  -> 32-row band
  const int waveN = wave & 1;           // 0..1  -> 64-col band
  const int bm    = blockIdx.x / NB;
  const int bn    = blockIdx.x % NB;

  const unsigned short* Atile = A  + (size_t)(bm * 128) * Kdim;
  const unsigned short* Btile = BT + (size_t)(bn * 128) * Kdim;
  const unsigned ldsA0 = (unsigned)(size_t)(&lsA[0][0]);
  const unsigned ldsA1 = (unsigned)(size_t)(&lsA[1][0]);
  const unsigned ldsB0 = (unsigned)(size_t)(&lsB[0][0]);
  const unsigned ldsB1 = (unsigned)(size_t)(&lsB[1][0]);

  v8f acc[2][4];
#pragma unroll
  for (int ms = 0; ms < 2; ++ms)
#pragma unroll
    for (int ns = 0; ns < 4; ++ns) acc[ms][ns] = (v8f){0.f,0.f,0.f,0.f,0.f,0.f,0.f,0.f};

  // prologue: TDM stages first K-slab, one wave issues for the workgroup
  if (wave == 0) {
    tdm_load_tile_2d(ldsA0, Atile, 32, 128, (unsigned)Kdim, 128, (unsigned)Kdim);
    tdm_load_tile_2d(ldsB0, Btile, 32, 128, (unsigned)Kdim, 128, (unsigned)Kdim);
    __builtin_amdgcn_s_wait_tensorcnt(0);
  }
  __syncthreads();

  for (int k0 = 0; k0 < Kdim; k0 += 32) {
    const int cur = (k0 >> 5) & 1;
    const bool more = (k0 + 32) < Kdim;
    if (wave == 0 && more) {           // prefetch next slab into other buffer
      tdm_load_tile_2d(cur ? ldsA0 : ldsA1, Atile + k0 + 32,
                       32, 128, (unsigned)Kdim, 128, (unsigned)Kdim);
      tdm_load_tile_2d(cur ? ldsB0 : ldsB1, Btile + k0 + 32,
                       32, 128, (unsigned)Kdim, 128, (unsigned)Kdim);
    }

    const unsigned short* la = lsA[cur];
    const unsigned short* lb = lsB[cur];
    FragAB af[2], bfm[4];
#pragma unroll
    for (int ms = 0; ms < 2; ++ms) {
      const unsigned short* ar = la + (waveM * 32 + ms * 16 + l16) * 32;
      af[ms].h[0] = *(const v8u*)(ar + half * 8);
      af[ms].h[1] = *(const v8u*)(ar + half * 8 + 16);
    }
#pragma unroll
    for (int ns = 0; ns < 4; ++ns) {
      const unsigned short* br = lb + (waveN * 64 + ns * 16 + l16) * 32;
      bfm[ns].h[0] = *(const v8u*)(br + half * 8);
      bfm[ns].h[1] = *(const v8u*)(br + half * 8 + 16);
    }
#pragma unroll
    for (int ms = 0; ms < 2; ++ms)
#pragma unroll
      for (int ns = 0; ns < 4; ++ns)
        acc[ms][ns] = __builtin_amdgcn_wmma_f32_16x16x32_bf16(
            false, af[ms].bf, false, bfm[ns].bf, (short)0, acc[ms][ns], false, false);

    if (wave == 0 && more) __builtin_amdgcn_s_wait_tensorcnt(0);
    __syncthreads();
  }

  // writeback
#pragma unroll
  for (int ms = 0; ms < 2; ++ms) {
#pragma unroll
    for (int ns = 0; ns < 4; ++ns) {
#pragma unroll
      for (int r = 0; r < 8; ++r) {
        const int row = bm * 128 + waveM * 32 + ms * 16 + r + 8 * half;
        const int col = bn * 128 + waveN * 64 + ns * 16 + l16;
        const float v = acc[ms][ns][r];
        if (mode == 3) {
          ((float*)out)[(size_t)row * (NB * 128) + col] = v;
        } else {
          const int bb = row / Nc, n = row % Nc;
          unsigned short* o = (unsigned short*)out;
          if (mode == 0) {
            const int hh = col / HDc, hd = col % HDc;
            o[(((size_t)bb * Hc + hh) * Nc + n) * HDc + hd] = f2bf(v);
          } else if (mode == 1) {
            const int kv = col / HDc, hd = col % HDc;
            o[(((size_t)bb * KVHc + kv) * Nc + n) * HDc + hd] = f2bf(v);
          } else {
            const int kv = col / HDc, hd = col % HDc;
            o[(((size_t)bb * KVHc + kv) * HDc + hd) * Nc + n] = f2bf(v);
          }
        }
      }
    }
  }
}

// ---------------- flash-attention kernel (one wave = 16 queries of one head) ----------------
__global__ void __launch_bounds__(256)
gqa_attn_kernel(const unsigned short* __restrict__ Q,   // [B][H][N][HD] bf16
                const unsigned short* __restrict__ K,   // [B][KVH][N][HD] bf16
                const unsigned short* __restrict__ VT,  // [B][KVH][HD][N] bf16
                unsigned short* __restrict__ O) {       // [B][N][H*HD] bf16
  __shared__ unsigned short pbuf[8][16 * 32];           // per-wave P tile (16q x 32k)

  const int wib  = threadIdx.x >> 5;
  const int lane = threadIdx.x & 31;
  const int half = lane >> 4;
  const int l16  = lane & 15;
  const int gwave = blockIdx.x * 8 + wib;

  const int qt  = gwave & (Nc / 16 - 1);     // 128 q-tiles per (b,h)
  const int bh  = gwave >> 7;
  const int h   = bh & (Hc - 1);
  const int b   = bh >> 4;
  const int kvh = h / NREP;                  // GQA: repeat_interleave -> h // N_REP
  const int q0  = qt * 16;

  const unsigned short* Qbase = Q  + ((size_t)(b * Hc + h) * Nc) * HDc;
  const unsigned short* Kbase = K  + ((size_t)(b * KVHc + kvh) * Nc) * HDc;
  const unsigned short* Vbase = VT + ((size_t)(b * KVHc + kvh) * HDc) * Nc;
  unsigned short* pl = pbuf[wib];

  // load Q fragments once: 4 chunks of K=32 over HD=128
  FragAB qf[4];
  {
    const unsigned short* qrow = Qbase + (size_t)(q0 + l16) * HDc;
#pragma unroll
    for (int c = 0; c < 4; ++c) {
      qf[c].h[0] = *(const v8u*)(qrow + c * 32 + half * 8);
      qf[c].h[1] = *(const v8u*)(qrow + c * 32 + half * 8 + 16);
    }
  }

  v8f oacc[8];
#pragma unroll
  for (int n = 0; n < 8; ++n) oacc[n] = (v8f){0.f,0.f,0.f,0.f,0.f,0.f,0.f,0.f};
  float m_[8], l_[8];
#pragma unroll
  for (int r = 0; r < 8; ++r) { m_[r] = -1e30f; l_[r] = 0.f; }

  const int kend = q0 + 16;                  // causal: keys < kend
  const int nkb  = (kend + 31) >> 5;         // key blocks of 32

  for (int kb = 0; kb < nkb; ++kb) {
    const int kbase = kb * 32;
    float S[2][8];

    __builtin_prefetch(Kbase + (size_t)(kbase + 32 + l16) * HDc, 0, 1);
    __builtin_prefetch(Vbase + (size_t)l16 * Nc + kbase + 32, 0, 1);

#pragma unroll
    for (int j = 0; j < 2; ++j) {
      const int kt = kbase + j * 16;
      v8f acc = {0.f,0.f,0.f,0.f,0.f,0.f,0.f,0.f};
      if (kt < kend) {
#pragma unroll
        for (int c = 0; c < 4; ++c) {
          FragAB kf;   // B operand: col = key, contraction over HD chunk
          const unsigned short* krow = Kbase + (size_t)(kt + l16) * HDc + c * 32 + half * 8;
          kf.h[0] = *(const v8u*)(krow);
          kf.h[1] = *(const v8u*)(krow + 16);
          acc = __builtin_amdgcn_wmma_f32_16x16x32_bf16(false, qf[c].bf, false, kf.bf,
                                                        (short)0, acc, false, false);
        }
      }
#pragma unroll
      for (int r = 0; r < 8; ++r) {
        const int qi = q0 + r + 8 * half;
        const int ki = kt + l16;
        S[j][r] = (kt < kend && ki <= qi) ? acc[r] * SCALEc : -1e30f;
      }
    }

    // online softmax over this 32-key block; row stats live in 16-lane halves
    float alpha[8];
#pragma unroll
    for (int r = 0; r < 8; ++r) {
      float bm = fmaxf(S[0][r], S[1][r]);
      for (int off = 1; off < 16; off <<= 1) bm = fmaxf(bm, __shfl_xor(bm, off, 32));
      const float nm = fmaxf(m_[r], bm);
      alpha[r] = __expf(m_[r] - nm);
      m_[r]    = nm;
      const float p0 = __expf(S[0][r] - nm);
      const float p1 = __expf(S[1][r] - nm);
      const int row  = r + 8 * half;
      pl[row * 32 + l16]      = f2bf(p0);
      pl[row * 32 + 16 + l16] = f2bf(p1);
      float ps = p0 + p1;
      for (int off = 1; off < 16; off <<= 1) ps += __shfl_xor(ps, off, 32);
      l_[r] = l_[r] * alpha[r] + ps;
    }
#pragma unroll
    for (int n = 0; n < 8; ++n)
#pragma unroll
      for (int r = 0; r < 8; ++r) oacc[n][r] *= alpha[r];

    // cross-lane C-layout -> A-layout for P goes through LDS; order explicitly
    asm volatile("s_wait_dscnt 0x0" ::: "memory");

    FragAB pf;   // A operand: row = local q, K = 32 keys of this block
    pf.h[0] = *(const v8u*)(pl + l16 * 32 + half * 8);
    pf.h[1] = *(const v8u*)(pl + l16 * 32 + half * 8 + 16);

    // O += P @ V  (8 WMMA over HD=128 in 16-col chunks; VT gives contiguous key axis)
#pragma unroll
    for (int n = 0; n < 8; ++n) {
      FragAB vf;
      const unsigned short* vrow = Vbase + (size_t)(n * 16 + l16) * Nc + kbase + half * 8;
      vf.h[0] = *(const v8u*)(vrow);
      vf.h[1] = *(const v8u*)(vrow + 16);
      oacc[n] = __builtin_amdgcn_wmma_f32_16x16x32_bf16(false, pf.bf, false, vf.bf,
                                                        (short)0, oacc[n], false, false);
    }
    asm volatile("s_wait_dscnt 0x0" ::: "memory");  // P loads done before next block's stores
  }

  // normalize and store attention output, row-major [B*N][H*HD] bf16
#pragma unroll
  for (int n = 0; n < 8; ++n) {
#pragma unroll
    for (int r = 0; r < 8; ++r) {
      const float v  = oacc[n][r] / l_[r];
      const int tok  = q0 + r + 8 * half;
      const int col  = h * HDc + n * 16 + l16;
      O[((size_t)b * Nc + tok) * (Hc * HDc) + col] = f2bf(v);
    }
  }
}

// ---------------- host launch ----------------
extern "C" void kernel_launch(void* const* d_in, const int* in_sizes, int n_in,
                              void* d_out, int out_size, void* d_ws, size_t ws_size,
                              hipStream_t stream) {
  (void)in_sizes; (void)n_in; (void)out_size; (void)ws_size;
  const float* x  = (const float*)d_in[0];
  // d_in[1] = causal mask, applied analytically (key <= query) -> not read
  const float* Wq = (const float*)d_in[2];
  const float* Wk = (const float*)d_in[3];
  const float* Wv = (const float*)d_in[4];
  const float* Wo = (const float*)d_in[5];

  char* ws = (char*)d_ws;
  size_t off = 0;
  auto alloc = [&](size_t bytes) {
    size_t o = off; off += (bytes + 255) & ~(size_t)255; return o;
  };
  // workspace need: ~80 MB
  unsigned short* xb  = (unsigned short*)(ws + alloc((size_t)Bc * Nc * DIMc * 2));
  unsigned short* wqt = (unsigned short*)(ws + alloc((size_t)DIMc * (Hc  * HDc) * 2));
  unsigned short* wkt = (unsigned short*)(ws + alloc((size_t)DIMc * (KVHc * HDc) * 2));
  unsigned short* wvt = (unsigned short*)(ws + alloc((size_t)DIMc * (KVHc * HDc) * 2));
  unsigned short* wot = (unsigned short*)(ws + alloc((size_t)(Hc * HDc) * DIMc * 2));
  unsigned short* qb  = (unsigned short*)(ws + alloc((size_t)Bc * Hc   * Nc * HDc * 2));
  unsigned short* kbf = (unsigned short*)(ws + alloc((size_t)Bc * KVHc * Nc * HDc * 2));
  unsigned short* vtb = (unsigned short*)(ws + alloc((size_t)Bc * KVHc * HDc * Nc * 2 + 256));
  unsigned short* ob  = (unsigned short*)(ws + alloc((size_t)Bc * Nc * Hc * HDc * 2));

  // 1) stage inputs as bf16 (weights transposed to [out][in])
  {
    size_t n = (size_t)Bc * Nc * DIMc;
    f32_to_bf16_kernel<<<(unsigned)((n + 255) / 256), 256, 0, stream>>>(x, xb, n);
  }
  transpose_f32_to_bf16_kernel<<<(DIMc * (Hc * HDc) + 255) / 256, 256, 0, stream>>>(Wq, wqt, DIMc, Hc * HDc);
  transpose_f32_to_bf16_kernel<<<(DIMc * (KVHc * HDc) + 255) / 256, 256, 0, stream>>>(Wk, wkt, DIMc, KVHc * HDc);
  transpose_f32_to_bf16_kernel<<<(DIMc * (KVHc * HDc) + 255) / 256, 256, 0, stream>>>(Wv, wvt, DIMc, KVHc * HDc);
  transpose_f32_to_bf16_kernel<<<((Hc * HDc) * DIMc + 255) / 256, 256, 0, stream>>>(Wo, wot, Hc * HDc, DIMc);

  const int MB = (Bc * Nc) / 128;           // 32 block-rows of the token dim

  // 2) projections via TDM-staged WMMA GEMM (block tile 128x128)
  wmma_gemm_kernel<<<MB * ((Hc  * HDc) / 128), 256, 0, stream>>>(xb, wqt, qb,  (Hc  * HDc) / 128, DIMc, 0);
  wmma_gemm_kernel<<<MB * ((KVHc * HDc) / 128), 256, 0, stream>>>(xb, wkt, kbf, (KVHc * HDc) / 128, DIMc, 1);
  wmma_gemm_kernel<<<MB * ((KVHc * HDc) / 128), 256, 0, stream>>>(xb, wvt, vtb, (KVHc * HDc) / 128, DIMc, 2);

  // 3) causal GQA flash attention
  gqa_attn_kernel<<<(Bc * Hc * (Nc / 16)) / 8, 256, 0, stream>>>(qb, kbf, vtb, ob);

  // 4) output projection -> f32
  wmma_gemm_kernel<<<MB * (DIMc / 128), 256, 0, stream>>>(ob, wot, d_out, DIMc / 128, Hc * HDc, 3);
}
